// GAT_42889543418178
// MI455X (gfx1250) — compile-verified, matrix operand order checked by reading, not run on hardware
//
#include <hip/hip_runtime.h>
#include <hip/hip_bf16.h>

// GAT: B=16, N=1024, F_IN=64, HID=64, OUT=64, NHEADS=4, TAXI=512, ALPHA=0.2
#define GB      16
#define GN      1024
#define GF      64
#define GH      64
#define GNH     4
#define GTAXI   512
#define GALPHA  0.2f

typedef __attribute__((ext_vector_type(16))) _Float16 v16h;
typedef __attribute__((ext_vector_type(8)))  _Float16 v8h;
typedef __attribute__((ext_vector_type(8)))  float    v8f;
typedef __attribute__((ext_vector_type(4)))  float    v4f;

// 16-bit A/B fragment K mapping (ISA 7.12.2): lane half selects K base 0/8,
// half-index j maps to K = 8*(lane>>4) + (j&7) + 16*(j>>3).
// => per lane the 16 halves are two contiguous 8-element K runs.

// A-fragment from row-major f32 [rows, ld]: 4 x b128 loads + convert.
__device__ __forceinline__ v16h load_a_frag(const float* __restrict__ src,
                                            int ld, int row0, int k0, int lane) {
    const float* p = src + (size_t)(row0 + (lane & 15)) * ld + k0 + ((lane >> 4) << 3);
    v4f q0 = *(const v4f*)(p);
    v4f q1 = *(const v4f*)(p + 4);
    v4f q2 = *(const v4f*)(p + 16);
    v4f q3 = *(const v4f*)(p + 20);
    v16h a;
#pragma unroll
    for (int i = 0; i < 4; ++i) {
        a[i]      = (_Float16)q0[i];
        a[4 + i]  = (_Float16)q1[i];
        a[8 + i]  = (_Float16)q2[i];
        a[12 + i] = (_Float16)q3[i];
    }
    return a;
}

// ---- Weight pre-pack: f32 [nmats][Kdim,64] -> f16 B-fragment layout ------
// One wave per fragment. blockIdx.x = (mat*(Kdim/32) + kstep)*4 + ss.
__global__ __launch_bounds__(32) void gat_pack_w(const float* __restrict__ W,
                                                 _Float16* __restrict__ out,
                                                 int Kdim) {
    const int lane  = threadIdx.x;
    const int ss    = blockIdx.x & 3;
    const int frag  = blockIdx.x >> 2;              // mat*(Kdim/32) + kstep
    const int ksteps = Kdim >> 5;
    const int kstep = frag % ksteps;
    const int mat   = frag / ksteps;
    const float* Wm = W + (size_t)mat * Kdim * GH;
    const int k0  = kstep * 32 + ((lane >> 4) << 3);
    const int col = ss * 16 + (lane & 15);
    v16h v;
#pragma unroll
    for (int j = 0; j < 16; ++j) {
        int k = k0 + (j & 7) + ((j >> 3) << 4);
        v[j] = (_Float16)Wm[(size_t)k * GH + col];
    }
    *(v16h*)(out + ((size_t)(blockIdx.x * 32 + lane) << 4)) = v;
}

// ---- Projection + coefficient kernel -------------------------------------
// Computes a 16x64 tile of h = x @ (node<TAXI ? Wa : Wb)[hd], then:
//  * stores h in f16 pre-packed WMMA-B-fragment layout (one v8h store per
//    subtile per lane: C-frag lane mapping == B-frag lane mapping)
//  * reduces A1 = h.a[:64], A2 = h.a[64:] across lanes via shfl_xor.
__global__ __launch_bounds__(32) void gat_proj(const float* __restrict__ xin, int ldx,
                                               const _Float16* __restrict__ wpA,
                                               const _Float16* __restrict__ wpB,
                                               const float* __restrict__ avec,
                                               _Float16* __restrict__ hpack,
                                               float* __restrict__ A1,
                                               float* __restrict__ A2, int Kdim) {
    const int hd   = blockIdx.y;
    const int lane = threadIdx.x;
    const int row0 = blockIdx.x * 16;              // global row in [0, B*N)
    const int bIdx = row0 >> 10;                   // row0 / GN
    const int nInB = row0 & (GN - 1);              // node index within batch
    const int ksteps = Kdim >> 5;
    const _Float16* wp = (nInB < GTAXI ? wpA : wpB)
                       + ((size_t)hd * ksteps * 4 * 32 << 4);

    v8f acc[4] = {};
    for (int k0 = 0; k0 < Kdim; k0 += 32) {
        v16h af = load_a_frag(xin, ldx, row0, k0, lane);
#pragma unroll
        for (int s = 0; s < 4; ++s) {
            v16h bf = *(const v16h*)(wp + ((size_t)(((k0 >> 5) * 4 + s) * 32 + lane) << 4));
            acc[s] = __builtin_amdgcn_wmma_f32_16x16x32_f16(
                false, af, false, bf, (short)0, acc[s], false, false);
        }
    }

    // --- pack h tile into B-fragment layout (f16) ---
    // C-frag: lane holds (row = 8*(lane>>4)+r, col = s*16+(lane&15)) == exactly
    // the 8 halves of B-frag lane `lane` for the 16-row half this tile covers.
    _Float16* hp = hpack + ((size_t)hd * GB + bIdx) * ((size_t)GN * GH);
    const int chunk = nInB >> 5;
    const int half  = (nInB >> 4) & 1;
#pragma unroll
    for (int s = 0; s < 4; ++s) {
        v8h v;
#pragma unroll
        for (int r = 0; r < 8; ++r) v[r] = (_Float16)acc[s][r];
        *(v8h*)(hp + ((size_t)((chunk * 4 + s) * 32 + lane) << 4) + half * 8) = v;
    }

    // --- A1/A2 via cross-lane reduction over the 16 columns per half ---
    const float* av   = avec + (size_t)hd * 2 * GH;
    const int    colb = lane & 15;
    const int    rOff = (lane >> 4) * 8;
    const size_t hb   = (size_t)hd * (GB * GN);
#pragma unroll
    for (int r = 0; r < 8; ++r) {
        float s1 = 0.f, s2 = 0.f;
#pragma unroll
        for (int s = 0; s < 4; ++s) {
            float v = acc[s][r];
            int   c = s * 16 + colb;
            s1 += v * av[c];
            s2 += v * av[GH + c];
        }
#pragma unroll
        for (int d = 1; d < 16; d <<= 1) {
            s1 += __shfl_xor(s1, d, 32);
            s2 += __shfl_xor(s2, d, 32);
        }
        if (colb == 0) {
            A1[hb + row0 + rOff + r] = s1;
            A2[hb + row0 + rOff + r] = s2;
        }
    }
}

// ---- Attention softmax + aggregation (flash-style, 16 rows per wave) ----
// e = mask * leaky_relu(A1[i]+A2[m]); attn = exp(e)/rowsum (no max shift:
// mathematically identical); dst = elu(attn @ h). Layer 2 also writes attn.
__global__ __launch_bounds__(32) void gat_attn(const _Float16* __restrict__ hpack,
                                               const float* __restrict__ A1,
                                               const float* __restrict__ A2,
                                               const float* __restrict__ mask,
                                               float* __restrict__ dst,
                                               int dstStride, int headColMul,
                                               float* __restrict__ attnOut) {
    const int hd   = blockIdx.y;
    const int lane = threadIdx.x;
    const int b    = blockIdx.x / (GN / 16);
    const int r0   = (blockIdx.x % (GN / 16)) * 16;   // row within batch

    const size_t headBase = (size_t)hd * (GB * GN);
    const _Float16* hp  = hpack + ((size_t)hd * GB + b) * ((size_t)GN * GH);
    const float*    A1b = A1 + headBase + (size_t)b * GN;
    const float*    A2b = A2 + headBase + (size_t)b * GN;

    const int   myrow = r0 + (lane & 15);
    const int   kb    = (lane >> 4) << 3;
    const float a1    = A1b[myrow];
    const float* mrow = mask + (size_t)myrow * GN;

    // pass 1: row sum of exp(e). Vectorized b128 loads of A2 and mask.
    float s = 0.f;
    for (int m0 = 0; m0 < GN; m0 += 32) {
        const int mA = m0 + kb;
        v4f a2q[4], mkq[4];
        a2q[0] = *(const v4f*)(A2b + mA);      a2q[1] = *(const v4f*)(A2b + mA + 4);
        a2q[2] = *(const v4f*)(A2b + mA + 16); a2q[3] = *(const v4f*)(A2b + mA + 20);
        mkq[0] = *(const v4f*)(mrow + mA);      mkq[1] = *(const v4f*)(mrow + mA + 4);
        mkq[2] = *(const v4f*)(mrow + mA + 16); mkq[3] = *(const v4f*)(mrow + mA + 20);
#pragma unroll
        for (int q = 0; q < 4; ++q)
#pragma unroll
            for (int i = 0; i < 4; ++i) {
                float e = a1 + a2q[q][i];
                e = e > 0.f ? e : GALPHA * e;
                e *= mkq[q][i];
                s += __expf(e);
            }
    }
    s += __shfl_xor(s, 16, 32);
    const float invs = 1.f / s;

    // pass 2: rebuild P fragment (A-matrix), WMMA against pre-packed h.
    v8f acc[4] = {};
    float* attnRow = attnOut ? attnOut + ((size_t)(b * GN + myrow)) * GN : nullptr;
    for (int m0 = 0; m0 < GN; m0 += 32) {
        const int mA = m0 + kb;
        if (m0 + 32 < GN)
            __builtin_prefetch(hp + ((size_t)(((m0 >> 5) + 1) * 4 * 32 + lane) << 4), 0, 1);
        v4f a2q[4], mkq[4];
        a2q[0] = *(const v4f*)(A2b + mA);      a2q[1] = *(const v4f*)(A2b + mA + 4);
        a2q[2] = *(const v4f*)(A2b + mA + 16); a2q[3] = *(const v4f*)(A2b + mA + 20);
        mkq[0] = *(const v4f*)(mrow + mA);      mkq[1] = *(const v4f*)(mrow + mA + 4);
        mkq[2] = *(const v4f*)(mrow + mA + 16); mkq[3] = *(const v4f*)(mrow + mA + 20);
        v16h pf;
#pragma unroll
        for (int q = 0; q < 4; ++q) {
            v4f pq;
#pragma unroll
            for (int i = 0; i < 4; ++i) {
                float e = a1 + a2q[q][i];
                e = e > 0.f ? e : GALPHA * e;
                e *= mkq[q][i];
                float p = __expf(e) * invs;
                pq[i] = p;
                pf[q * 4 + i] = (_Float16)p;
            }
            if (attnRow)   // each (row,m) covered exactly once by A-frag layout
                *(v4f*)(attnRow + mA + (q >> 1) * 16 + (q & 1) * 4) = pq;
        }
#pragma unroll
        for (int ss = 0; ss < 4; ++ss) {
            v16h bf = *(const v16h*)(hp + ((size_t)(((m0 >> 5) * 4 + ss) * 32 + lane) << 4));
            acc[ss] = __builtin_amdgcn_wmma_f32_16x16x32_f16(
                false, pf, false, bf, (short)0, acc[ss], false, false);
        }
    }

    // ELU + writeback
    const int colb = lane & 15;
    const int rOff = (lane >> 4) * 8;
    const int cOff = hd * headColMul;
#pragma unroll
    for (int ss = 0; ss < 4; ++ss)
#pragma unroll
        for (int r = 0; r < 8; ++r) {
            float v = acc[ss][r];
            v = v > 0.f ? v : (__expf(v) - 1.f);
            int row = r0 + rOff + r;
            dst[((size_t)(b * GN + row)) * dstStride + cOff + ss * 16 + colb] = v;
        }
}

extern "C" void kernel_launch(void* const* d_in, const int* in_sizes, int n_in,
                              void* d_out, int out_size, void* d_ws, size_t ws_size,
                              hipStream_t stream) {
    const float* x    = (const float*)d_in[0];  // [B,N,64]
    const float* mask = (const float*)d_in[1];  // [N,N]
    const float* W1   = (const float*)d_in[2];  // [4,64,64]
    const float* W2   = (const float*)d_in[3];  // [4,64,64]
    const float* a    = (const float*)d_in[4];  // [4,128]
    const float* W1o  = (const float*)d_in[5];  // [256,64]
    const float* W2o  = (const float*)d_in[6];  // [256,64]
    const float* ao   = (const float*)d_in[7];  // [128]

    float* out     = (float*)d_out;                       // [B,N,64]
    float* attnOut = out + (size_t)GB * GN * GH;          // [B,N,N]

    // workspace carve-up
    float* ws   = (float*)d_ws;
    float* xc   = ws;                                     // [B*N][256] f32
    float* A1_1 = xc + (size_t)GB * GN * (GNH * GH);      // [4*B*N]
    float* A2_1 = A1_1 + (size_t)GNH * GB * GN;
    float* A1o  = A2_1 + (size_t)GNH * GB * GN;           // [B*N]
    float* A2o  = A1o + (size_t)GB * GN;
    _Float16* hpack1 = (_Float16*)(A2o + (size_t)GB * GN);    // [4][B][N*64] f16
    _Float16* hpack2 = hpack1 + (size_t)GNH * GB * GN * GH;   // [B][N*64] f16
    _Float16* wpW1   = hpack2 + (size_t)GB * GN * GH;         // [4][2][4][32][16]
    _Float16* wpW2   = wpW1 + (size_t)GNH * 2 * 4 * 32 * 16;
    _Float16* wpW1o  = wpW2 + (size_t)GNH * 2 * 4 * 32 * 16;  // [8][4][32][16]
    _Float16* wpW2o  = wpW1o + (size_t)8 * 4 * 32 * 16;

    const int tiles = GB * GN / 16;

    // ---- pre-pack weights into WMMA B-fragment f16 layout ----
    gat_pack_w<<<GNH * (GF / 32) * 4, 32, 0, stream>>>(W1, wpW1, GF);
    gat_pack_w<<<GNH * (GF / 32) * 4, 32, 0, stream>>>(W2, wpW2, GF);
    gat_pack_w<<<(GNH * GH / 32) * 4, 32, 0, stream>>>(W1o, wpW1o, GNH * GH);
    gat_pack_w<<<(GNH * GH / 32) * 4, 32, 0, stream>>>(W2o, wpW2o, GNH * GH);

    // ---- layer 1 ----
    gat_proj<<<dim3(tiles, GNH), 32, 0, stream>>>(
        x, GF, wpW1, wpW2, a, hpack1, A1_1, A2_1, GF);
    gat_attn<<<dim3(tiles, GNH), 32, 0, stream>>>(
        hpack1, A1_1, A2_1, mask, xc, GNH * GH, GH, nullptr);

    // ---- layer 2 ----
    gat_proj<<<dim3(tiles, 1), 32, 0, stream>>>(
        xc, GNH * GH, wpW1o, wpW2o, ao, hpack2, A1o, A2o, GNH * GH);
    gat_attn<<<dim3(tiles, 1), 32, 0, stream>>>(
        hpack2, A1o, A2o, mask, out, GH, 0, attnOut);
}
